// Encoder_68925635166714
// MI455X (gfx1250) — matrix operand order repaired and checked by reading, not compiled, not run on hardware
//
#include <hip/hip_runtime.h>
#include <math.h>
#include <stdint.h>

// ---------------- problem constants ----------------
constexpr int B_  = 32;
constexpr int T_  = 12;
constexpr int N_  = 512;
constexpr int C_  = 2;
constexpr int H_  = 64;
constexpr int K_  = 3;
constexpr int P_  = C_ + H_;   // 66
constexpr int PP_ = 80;        // padded per-k width (5 tiles of 16)
constexpr int Q_  = 256;       // padded K*PP (240 -> 256), 8 k-tiles of 32
constexpr int M_  = B_ * N_;   // 16384 rows in GEMM2

// ---------------- WMMA vector types ----------------
typedef __attribute__((ext_vector_type(16))) __bf16 bf16x16;
typedef __attribute__((ext_vector_type(8)))  __bf16 bf16x8;
typedef __attribute__((ext_vector_type(8)))  float  f32x8;

__device__ __forceinline__ bf16x16 load_frag(const __bf16* p0, const __bf16* p1) {
    union { bf16x16 v; bf16x8 h[2]; } u;
    u.h[0] = *(const bf16x8*)p0;   // 16B aligned by construction
    u.h[1] = *(const bf16x8*)p1;
    return u.v;
}

__device__ __forceinline__ f32x8 wmma_bf16(bf16x16 a, bf16x16 b, f32x8 c) {
    return __builtin_amdgcn_wmma_f32_16x16x32_bf16(
        false, a, false, b, (short)0, c, false, false);
}

__device__ __forceinline__ float sigmoidf_(float x) {
    return 1.0f / (1.0f + __expf(-x));
}

// ---------------- setup kernels ----------------
__global__ void k_cvt_g(const float* __restrict__ G, __bf16* __restrict__ Gbf, int n) {
    int i = blockIdx.x * blockDim.x + threadIdx.x;
    if (i < n) Gbf[i] = (__bf16)G[i];
}

// WgT: (128, 256) bf16, WuT: (64, 256) bf16, transposed + padded (zeros in pad)
__global__ void k_prep_w(const float* __restrict__ Wg, const float* __restrict__ Wu,
                         __bf16* __restrict__ WgT, __bf16* __restrict__ WuT) {
    int i = blockIdx.x * blockDim.x + threadIdx.x;   // (row, q): 192*256
    if (i >= 192 * Q_) return;
    int q   = i & (Q_ - 1);
    int row = i >> 8;
    int k = q / PP_, p = q % PP_;
    bool valid = (q < K_ * PP_) && (p < P_);
    float v = 0.0f;
    if (row < 2 * H_) {
        if (valid) v = Wg[(k * P_ + p) * (2 * H_) + row];
        WgT[row * Q_ + q] = (__bf16)v;
    } else {
        int n = row - 2 * H_;
        if (valid) v = Wu[(k * P_ + p) * H_ + n];
        WuT[n * Q_ + q] = (__bf16)v;
    }
}

__global__ void k_copy_f32(const float* __restrict__ src, float* __restrict__ dst, int n) {
    int i = blockIdx.x * blockDim.x + threadIdx.x;
    if (i < n) dst[i] = src[i];
}

// zero sup columns 240..255 once (never written by GEMM1)
__global__ void k_zero_pad(__bf16* __restrict__ sup) {
    int i = blockIdx.x * blockDim.x + threadIdx.x;   // M_*16
    if (i >= M_ * 16) return;
    int r = i >> 4, q = 240 + (i & 15);
    sup[(size_t)r * Q_ + q] = (__bf16)0.0f;
}

// ---------------- per-step elementwise kernels ----------------
// combT layout: (B, PP_, N) bf16 : comb[b][p][j]
__global__ void k_comb1(const float* __restrict__ x_seq, const float* __restrict__ h,
                        __bf16* __restrict__ comb, int t) {
    int i = blockIdx.x * blockDim.x + threadIdx.x;   // B*PP_*N
    if (i >= B_ * PP_ * N_) return;
    int j = i & (N_ - 1);
    int p = (i / N_) % PP_;
    int b = i / (N_ * PP_);
    float v = 0.0f;
    if (p < C_)      v = x_seq[(((size_t)(b * T_ + t) * N_ + j) * C_) + p];
    else if (p < P_) v = h[((size_t)b * N_ + j) * H_ + (p - C_)];
    comb[i] = (__bf16)v;
}

__global__ void k_comb2(const float* __restrict__ x_seq, const float* __restrict__ h,
                        const float* __restrict__ zr, __bf16* __restrict__ comb, int t) {
    int i = blockIdx.x * blockDim.x + threadIdx.x;
    if (i >= B_ * PP_ * N_) return;
    int j = i & (N_ - 1);
    int p = (i / N_) % PP_;
    int b = i / (N_ * PP_);
    float v = 0.0f;
    if (p < C_) {
        v = x_seq[(((size_t)(b * T_ + t) * N_ + j) * C_) + p];
    } else if (p < P_) {
        int hh = p - C_;
        float r = zr[((size_t)b * N_ + j) * (2 * H_) + H_ + hh];
        v = r * h[((size_t)b * N_ + j) * H_ + hh];
    }
    comb[i] = (__bf16)v;
}

// ---------------- GEMM1: sup[b, i, k*80+p] = sum_j G[k,i,j] * comb[b,p,j] ----------------
// block = 8 waves sharing one b; each wave one (k, i-tile-pair) combo (two 16-row
// A tiles -> 10 accumulators, each LDS B fragment feeds 2 WMMAs).
// comb j-chunk (80 x 32 bf16 = 5KB) double-buffered in LDS; stage for chunk jt+1
// is issued before computing chunk jt so global latency overlaps WMMA work.
// grid: B_ * 6 blocks (6 blocks x 8 waves cover K_*16 = 48 combos per b)
__global__ void k_gemm_g(const __bf16* __restrict__ Gbf, const __bf16* __restrict__ comb,
                         __bf16* __restrict__ sup) {
    __shared__ __align__(16) uint32_t sComb[2][PP_ * 16];   // 2 x 5KB

    int tid  = threadIdx.x;
    int wid  = tid >> 5;
    int lane = tid & 31;
    int b    = blockIdx.x / 6;
    int g    = blockIdx.x % 6;
    int combo = g * 8 + wid;                        // 0..47
    int k   = combo >> 4;                           // 0..2
    int it2 = combo & 15;                           // i-tile pair
    int i0  = it2 * 32;

    const __bf16* A  = Gbf + (size_t)k * N_ * N_;
    const uint4*  cb = (const uint4*)(comb + (size_t)b * PP_ * N_);  // row = 4 uint4

    // staging map: uint4 index i (0..319): row p = i>>2, 16B-group (i&3) within chunk row
    bool  has2 = tid < 64;
    uint4 r0, r1;
    auto issue = [&](int jt) {
        int base = jt * 8;                           // chunk offset in uint4 (32 bf16 = 8 uint4... per row: 4)
        // comb row stride = 512 bf16 = 64 uint4; chunk column offset = jt*32 bf16 = jt*4 uint4
        int i = tid;
        r0 = cb[(i >> 2) * 64 + jt * 4 + (i & 3)];
        if (has2) {
            int i2 = tid + 256;
            r1 = cb[(i2 >> 2) * 64 + jt * 4 + (i2 & 3)];
        }
        (void)base;
    };

    int arow0  = i0 + (lane & 15);
    int arow1  = arow0 + 16;
    int achunk = (lane >> 4) * 8;       // A: K-offsets achunk..+7 and achunk+16..+23
    int bhalf  = (lane >> 4) * 8;       // B: K-offsets bhalf..+7 and bhalf+16..+23
    int bp     = lane & 15;             // B: output column within tile

    f32x8 acc0[5] = {};
    f32x8 acc1[5] = {};
    issue(0);
    int cur = 0;
    for (int jt = 0; jt < 16; ++jt) {
        // store staged regs for this chunk
        *(uint4*)&sComb[cur][(size_t)tid * 4] = r0;
        if (has2) *(uint4*)&sComb[cur][(size_t)(tid + 256) * 4] = r1;
        __syncthreads();
        if (jt < 15) issue(jt + 1);      // overlap next stage with compute

        int j0 = jt * 32;
        const __bf16* ap0 = A + (size_t)arow0 * N_ + j0 + achunk;
        const __bf16* ap1 = A + (size_t)arow1 * N_ + j0 + achunk;
        bf16x16 a0 = load_frag(ap0, ap0 + 16);
        bf16x16 a1 = load_frag(ap1, ap1 + 16);
        const __bf16* sbase = (const __bf16*)sComb[cur];
#pragma unroll
        for (int pt = 0; pt < 5; ++pt) {
            const __bf16* bpp = sbase + (pt * 16 + bp) * 32 + bhalf;
            bf16x16 bfrag = load_frag(bpp, bpp + 16);
            acc0[pt] = wmma_bf16(a0, bfrag, acc0[pt]);
            acc1[pt] = wmma_bf16(a1, bfrag, acc1[pt]);
        }
        cur ^= 1;
    }
    // D layout: lane holds column (lane&15), rows v + ((lane&16)?8:0)
    int mb = (lane & 16) >> 1;
    int nc = lane & 15;
    size_t rowbase = (size_t)b * N_ + i0;
#pragma unroll
    for (int pt = 0; pt < 5; ++pt) {
#pragma unroll
        for (int v = 0; v < 8; ++v) {
            sup[(rowbase + mb + v) * Q_ + k * PP_ + pt * 16 + nc]      = (__bf16)acc0[pt][v];
            sup[(rowbase + 16 + mb + v) * Q_ + k * PP_ + pt * 16 + nc] = (__bf16)acc1[pt][v];
        }
    }
}

// ---------------- GEMM2 gate: zr = sigmoid(sup @ WgT^T + bg), (M,128) ----------------
// one wave per row-tile (1024 waves), 8 n-tiles per wave; WgT (64KB) staged in LDS
__global__ void k_gemm_gate(const __bf16* __restrict__ sup, const __bf16* __restrict__ WgT,
                            const float* __restrict__ bg, float* __restrict__ zr) {
    __shared__ uint32_t sW[2 * H_ * Q_ / 2];        // 16384 dwords = 64KB

    const uint32_t* wsrc = (const uint32_t*)WgT;
#pragma unroll
    for (int i = 0; i < 64; ++i)                    // cooperative stage of full WgT
        sW[threadIdx.x + i * 256] = wsrc[threadIdx.x + i * 256];
    __syncthreads();

    int wave = (blockIdx.x * blockDim.x + threadIdx.x) >> 5;
    int lane = threadIdx.x & 31;
    int R0     = wave * 16;
    int arow   = R0 + (lane & 15);
    int achunk = (lane >> 4) * 8;
    int bhalf  = (lane >> 4) * 8;
    int bn     = lane & 15;
    const __bf16* sw = (const __bf16*)sW;

    f32x8 acc[8] = {};
    for (int qt = 0; qt < 8; ++qt) {
        int q0 = qt * 32;
        const __bf16* ap = sup + (size_t)arow * Q_ + q0 + achunk;
        bf16x16 afrag = load_frag(ap, ap + 16);
#pragma unroll
        for (int nt = 0; nt < 8; ++nt) {
            const __bf16* wp = sw + (size_t)(nt * 16 + bn) * Q_ + q0 + bhalf;
            bf16x16 bfrag = load_frag(wp, wp + 16);
            acc[nt] = wmma_bf16(afrag, bfrag, acc[nt]);
        }
    }
    int mb = (lane & 16) >> 1;
#pragma unroll
    for (int nt = 0; nt < 8; ++nt) {
        int n = nt * 16 + bn;
        float bias = bg[n];
#pragma unroll
        for (int v = 0; v < 8; ++v) {
            zr[(size_t)(R0 + mb + v) * (2 * H_) + n] = sigmoidf_(acc[nt][v] + bias);
        }
    }
}

// ---------------- GEMM2 update: hc = tanh(sup @ WuT^T + bu); h = (1-z)h + z*hc ------
// WuT (32KB) staged in LDS
__global__ void k_gemm_upd(const __bf16* __restrict__ sup, const __bf16* __restrict__ WuT,
                           const float* __restrict__ bu, const float* __restrict__ zr,
                           const float* __restrict__ h_in, float* __restrict__ h_out) {
    __shared__ uint32_t sW[H_ * Q_ / 2];            // 8192 dwords = 32KB

    const uint32_t* wsrc = (const uint32_t*)WuT;
#pragma unroll
    for (int i = 0; i < 32; ++i)
        sW[threadIdx.x + i * 256] = wsrc[threadIdx.x + i * 256];
    __syncthreads();

    int wave = (blockIdx.x * blockDim.x + threadIdx.x) >> 5;
    int lane = threadIdx.x & 31;
    int R0     = wave * 16;
    int arow   = R0 + (lane & 15);
    int achunk = (lane >> 4) * 8;
    int bhalf  = (lane >> 4) * 8;
    int bn     = lane & 15;
    const __bf16* sw = (const __bf16*)sW;

    f32x8 acc[4] = {};
    for (int qt = 0; qt < 8; ++qt) {
        int q0 = qt * 32;
        const __bf16* ap = sup + (size_t)arow * Q_ + q0 + achunk;
        bf16x16 afrag = load_frag(ap, ap + 16);
#pragma unroll
        for (int nt = 0; nt < 4; ++nt) {
            const __bf16* wp = sw + (size_t)(nt * 16 + bn) * Q_ + q0 + bhalf;
            bf16x16 bfrag = load_frag(wp, wp + 16);
            acc[nt] = wmma_bf16(afrag, bfrag, acc[nt]);
        }
    }
    int mb = (lane & 16) >> 1;
#pragma unroll
    for (int nt = 0; nt < 4; ++nt) {
        int n = nt * 16 + bn;
        float bias = bu[n];
#pragma unroll
        for (int v = 0; v < 8; ++v) {
            size_t R = (size_t)(R0 + mb + v);
            float hc = tanhf(acc[nt][v] + bias);
            float z  = zr[R * (2 * H_) + n];
            float ho = h_in[R * H_ + n];
            h_out[R * H_ + n] = (1.0f - z) * ho + z * hc;
        }
    }
}

// ---------------- host launch ----------------
extern "C" void kernel_launch(void* const* d_in, const int* in_sizes, int n_in,
                              void* d_out, int out_size, void* d_ws, size_t ws_size,
                              hipStream_t stream) {
    const float* G      = (const float*)d_in[0];
    const float* x_seq  = (const float*)d_in[1];
    const float* init_h = (const float*)d_in[2];
    const float* Wg     = (const float*)d_in[3];
    const float* bg     = (const float*)d_in[4];
    const float* Wu     = (const float*)d_in[5];
    const float* bu     = (const float*)d_in[6];
    float* out = (float*)d_out;

    char* ws = (char*)d_ws;
    size_t off = 0;
    auto carve = [&](size_t bytes) {
        void* p = ws + off;
        off = (off + bytes + 255) & ~(size_t)255;
        return p;
    };
    __bf16* Gbf  = (__bf16*)carve((size_t)K_ * N_ * N_ * 2);   // 1.5 MB
    __bf16* WgT  = (__bf16*)carve((size_t)2 * H_ * Q_ * 2);    // 64 KB
    __bf16* WuT  = (__bf16*)carve((size_t)H_ * Q_ * 2);        // 32 KB
    __bf16* comb = (__bf16*)carve((size_t)B_ * PP_ * N_ * 2);  // 2.5 MB
    __bf16* sup  = (__bf16*)carve((size_t)M_ * Q_ * 2);        // 8 MB
    float*  zr   = (float*)carve((size_t)M_ * 2 * H_ * 4);     // 8 MB
    float*  h    = (float*)carve((size_t)M_ * H_ * 4);         // 4 MB

    const int TB = 256;
    // setup
    k_cvt_g   <<<(K_ * N_ * N_ + TB - 1) / TB, TB, 0, stream>>>(G, Gbf, K_ * N_ * N_);
    k_prep_w  <<<(192 * Q_ + TB - 1) / TB,     TB, 0, stream>>>(Wg, Wu, WgT, WuT);
    k_copy_f32<<<(M_ * H_ + TB - 1) / TB,      TB, 0, stream>>>(init_h, h, M_ * H_);
    k_zero_pad<<<(M_ * 16 + TB - 1) / TB,      TB, 0, stream>>>(sup);

    const int comb_blocks  = (B_ * PP_ * N_ + TB - 1) / TB;  // 5120
    const int gemm1_blocks = B_ * 6;                         // 192 blocks, 8 waves each
    const int gemm2_blocks = ((M_ / 16) * 32) / TB;          // 1024 waves -> 128 blocks

    for (int t = 0; t < T_; ++t) {
        k_comb1    <<<comb_blocks,  TB, 0, stream>>>(x_seq, h, comb, t);
        k_gemm_g   <<<gemm1_blocks, TB, 0, stream>>>(Gbf, comb, sup);
        k_gemm_gate<<<gemm2_blocks, TB, 0, stream>>>(sup, WgT, bg, zr);
        k_comb2    <<<comb_blocks,  TB, 0, stream>>>(x_seq, h, zr, comb, t);
        k_gemm_g   <<<gemm1_blocks, TB, 0, stream>>>(Gbf, comb, sup);
        float* h_out = (t == T_ - 1) ? out : h;
        k_gemm_upd <<<gemm2_blocks, TB, 0, stream>>>(sup, WuT, bu, zr, h, h_out);
    }
}
